// PSA_3882650436039
// MI455X (gfx1250) — matrix operand order repaired and checked by reading, not compiled, not run on hardware
//
#include <hip/hip_runtime.h>
#include <math.h>

typedef __attribute__((ext_vector_type(16))) _Float16 v16h;
typedef __attribute__((ext_vector_type(4)))  _Float16 v4h;
typedef __attribute__((ext_vector_type(8)))  float    v8f;

#define N_IMG 4
#define C_IN 2048
#define HFULL 59
#define WFULL 59
#define HWFULL (HFULL*WFULL)   // 3481
#define MIDC 512
#define HS 30
#define WS 30
#define PIX (HS*WS)            // 900
#define MMASK (59*59)          // 3481
#define KPAD 928               // 29*32 zero-padded K for attention GEMM
#define NTPW 4                 // N-tiles per wave (16x64 strip)

// ---------------- WMMA fragment loaders (ISA 7.12.2 layouts) ----------------
// A 16x32 f16: lanes 0-15 row M=lane, halves j<8 -> K=k+j, j>=8 -> K=k+16+(j-8);
// lanes 16-31 same rows, K offset +8.  All loads 8B-aligned (lda multiple of 4).
__device__ __forceinline__ v16h load_a_frag(const _Float16* A, long long rowBase, int k, int lane) {
  int off = (lane & 16) ? 8 : 0;
  const _Float16* p = A + rowBase + k + off;
  v4h c0 = *(const v4h*)(p);
  v4h c1 = *(const v4h*)(p + 4);
  v4h c2 = *(const v4h*)(p + 16);
  v4h c3 = *(const v4h*)(p + 20);
  v16h r;
#pragma unroll
  for (int t = 0; t < 4; ++t) { r[t] = c0[t]; r[4+t] = c1[t]; r[8+t] = c2[t]; r[12+t] = c3[t]; }
  return r;
}
// B 32x16 f16 col-major: col N=lane%16, halves j<8 -> K=k+j, j>=8 -> K=k+16+(j-8); lanes 16-31 K offset +8.
__device__ __forceinline__ v16h load_b_frag(const _Float16* B, long long colBase, int k, int lane) {
  int off = (lane & 16) ? 8 : 0;
  const _Float16* p = B + colBase + k + off;
  v4h c0 = *(const v4h*)(p);
  v4h c1 = *(const v4h*)(p + 4);
  v4h c2 = *(const v4h*)(p + 16);
  v4h c3 = *(const v4h*)(p + 20);
  v16h r;
#pragma unroll
  for (int t = 0; t < 4; ++t) { r[t] = c0[t]; r[4+t] = c1[t]; r[8+t] = c2[t]; r[12+t] = c3[t]; }
  return r;
}

// ---------------- Generic batched WMMA GEMM with fused BN+ReLU --------------
// C[M x Ncols] = act(bn(A[M x K] * B[K x Ncols]))
// A row-major fp16 (stride lda), B col-major fp16 (col stride ldb, ldb%32==0,
// rows >= true-K must be zero in B).  Each wave computes a 16(M) x 64(N) strip.
// K loop is a 2x-unrolled ping-pong pipeline: buffer sets alternate roles, so
// loads stay one step ahead of the WMMAs with zero register-rotation moves.
__global__ __launch_bounds__(256)
void gemm_wmma_kernel(const _Float16* __restrict__ A, const _Float16* __restrict__ B,
                      float* out32, _Float16* out16c, _Float16* out16r,
                      const float* bnp, int doRelu,
                      int M, int Ncols, int lda, int ldb, int kSteps,
                      long long aBatch, long long bBatch,
                      int ld32, long long o32Batch,
                      int ldoc, long long ocBatch,
                      int ldor, long long orBatch) {
  int lane = threadIdx.x & 31;
  int wave = threadIdx.x >> 5;
  int tm = blockIdx.x;
  int tn0 = (blockIdx.y * 8 + wave) * NTPW;         // first of 4 N-tiles for this wave
  int nT = (Ncols + 15) >> 4;
  if (tm * 16 >= M || tn0 >= nT) return;            // uniform per wave: EXEC stays all-ones
  int nb = blockIdx.z;

  const _Float16* Ab = A + (long long)nb * aBatch;
  const _Float16* Bb = B + (long long)nb * bBatch;

  int rowA = tm * 16 + (lane & 15); if (rowA >= M) rowA = M - 1;   // clamp, stores guarded
  long long aRowBase = (long long)rowA * lda;

  long long bColBase[NTPW];
#pragma unroll
  for (int t = 0; t < NTPW; ++t) {
    int colB = (tn0 + t) * 16 + (lane & 15);
    if (colB >= Ncols) colB = Ncols - 1;                           // clamp, stores guarded
    bColBase[t] = (long long)colB * ldb;
  }

  v8f acc[NTPW];
#pragma unroll
  for (int t = 0; t < NTPW; ++t) acc[t] = (v8f){};

#define LOAD_FRAGS(aF, bF, k)                                              \
  do {                                                                     \
    aF = load_a_frag(Ab, aRowBase, (k), lane);                             \
    _Pragma("unroll")                                                      \
    for (int t = 0; t < NTPW; ++t) bF[t] = load_b_frag(Bb, bColBase[t], (k), lane); \
  } while (0)

#define DO_WMMA(aF, bF)                                                    \
  do {                                                                     \
    acc[0] = __builtin_amdgcn_wmma_f32_16x16x32_f16(false, aF, false, bF[0], (short)0, acc[0], false, false); \
    acc[1] = __builtin_amdgcn_wmma_f32_16x16x32_f16(false, aF, false, bF[1], (short)0, acc[1], false, false); \
    acc[2] = __builtin_amdgcn_wmma_f32_16x16x32_f16(false, aF, false, bF[2], (short)0, acc[2], false, false); \
    acc[3] = __builtin_amdgcn_wmma_f32_16x16x32_f16(false, aF, false, bF[3], (short)0, acc[3], false, false); \
  } while (0)

  // ---- 2x-unrolled ping-pong software pipeline over k (kSteps is uniform) ----
  v16h a0, a1, b0[NTPW], b1[NTPW];
  LOAD_FRAGS(a0, b0, 0);
  int ks = 0;
  for (; ks + 2 < kSteps; ks += 2) {
    LOAD_FRAGS(a1, b1, (ks + 1) * 32);   // lookahead loads overlap wmma(buf0)
    DO_WMMA(a0, b0);
    LOAD_FRAGS(a0, b0, (ks + 2) * 32);   // lookahead loads overlap wmma(buf1)
    DO_WMMA(a1, b1);
  }
  if (ks + 1 < kSteps) {                 // 2 steps remain
    LOAD_FRAGS(a1, b1, (ks + 1) * 32);
    DO_WMMA(a0, b0);
    DO_WMMA(a1, b1);
  } else {                               // 1 step remains
    DO_WMMA(a0, b0);
  }
#undef LOAD_FRAGS
#undef DO_WMMA

  // fused per-row BN (+ReLU) applied once per output row of this lane
  int mBase = tm * 16 + ((lane & 16) ? 8 : 0);      // C/D layout: VGPR r -> M = r (+8 hi lanes)
  float sRow[8], tRow[8];
#pragma unroll
  for (int r = 0; r < 8; ++r) {
    int mg = mBase + r; if (mg >= M) mg = M - 1;
    if (bnp) {  // bn params stacked (4, M): g, b, mean, var
      float g = bnp[mg], bb = bnp[M + mg], mn = bnp[2 * M + mg], vr = bnp[3 * M + mg];
      float s = g * rsqrtf(vr + 1e-5f);
      sRow[r] = s; tRow[r] = bb - mn * s;
    } else { sRow[r] = 1.f; tRow[r] = 0.f; }
  }

#pragma unroll
  for (int t = 0; t < NTPW; ++t) {
    int nOut = (tn0 + t) * 16 + (lane & 15);
    if (nOut >= Ncols) continue;
#pragma unroll
    for (int r = 0; r < 8; ++r) {
      int mg = mBase + r;
      if (mg >= M) continue;
      float v = acc[t][r] * sRow[r] + tRow[r];
      if (doRelu) v = fmaxf(v, 0.f);
      if (out32)  out32 [nb * o32Batch + (long long)mg * ld32 + nOut] = v;
      if (out16c) out16c[nb * ocBatch + (long long)nOut * ldoc + mg] = (_Float16)v;
      if (out16r) out16r[nb * orBatch + (long long)mg * ldor + nOut] = (_Float16)v;
    }
  }
}

// ---------------- small helper kernels --------------------------------------
__global__ void f32_to_f16_kernel(const float* __restrict__ src, _Float16* __restrict__ dst, long long count) {
  long long i = (long long)blockIdx.x * blockDim.x + threadIdx.x;
  if (i < count) dst[i] = (_Float16)src[i];
}

// Subsample x at (2i,2j) (exact align-corners 59->30 resize) into fp16
// col-major activations: xs[(n*900+p)*2048 + c].
__global__ void pack_x_kernel(const float* __restrict__ x, _Float16* __restrict__ xs) {
  long long idx = (long long)blockIdx.x * blockDim.x + threadIdx.x;
  const long long total = (long long)N_IMG * PIX * C_IN;
  if (idx >= total) return;
  int c = (int)(idx % C_IN);
  int p = (int)((idx / C_IN) % PIX);
  int n = (int)(idx / ((long long)C_IN * PIX));
  int i = p / WS, j = p % WS;
  float v = x[(((long long)n * C_IN + c) * HFULL + 2 * i) * WFULL + 2 * j];
  xs[idx] = (_Float16)v;
}

// PSA mask gather + softmax over the 900 "source" positions for one output
// column j; writes column of ym (col stride KPAD, rows 900..927 zeroed).
// COLLECT: v_i = Y[(ia-ja+29)*59 + (ib-jb+29), j]   (column-local gather)
// DISTRIB: v_i = Y[(ja-ia+29)*59 + (jb-ib+29), i]
template <int COLLECT>
__global__ __launch_bounds__(256)
void psa_softmax_kernel(const float* __restrict__ Y, _Float16* __restrict__ ym) {
  int j = blockIdx.x;
  int n = blockIdx.y;
  int tid = threadIdx.x;
  const float* Yn = Y + (long long)n * MMASK * PIX;
  _Float16* ymn = ym + (long long)n * PIX * KPAD + (long long)j * KPAD;
  int ja = j / WS, jb = j % WS;
  float v[4];
  float lmax = -3.4e38f;
#pragma unroll
  for (int t = 0; t < 4; ++t) {
    int i = tid + t * 256;
    float val = -3.4e38f;
    if (i < PIX) {
      int ia = i / WS, ib = i % WS;
      int r, col;
      if (COLLECT) { r = (ia - ja + 29) * 59 + (ib - jb + 29); col = j; }
      else         { r = (ja - ia + 29) * 59 + (jb - ib + 29); col = i; }
      val = Yn[(long long)r * PIX + col];
    }
    v[t] = val;
    lmax = fmaxf(lmax, val);
  }
  __shared__ float red[256];
  red[tid] = lmax; __syncthreads();
  for (int s = 128; s > 0; s >>= 1) { if (tid < s) red[tid] = fmaxf(red[tid], red[tid + s]); __syncthreads(); }
  float mx = red[0]; __syncthreads();
  float lsum = 0.f;
#pragma unroll
  for (int t = 0; t < 4; ++t) {
    int i = tid + t * 256;
    float e = (i < PIX) ? expf(v[t] - mx) : 0.f;
    v[t] = e; lsum += e;
  }
  red[tid] = lsum; __syncthreads();
  for (int s = 128; s > 0; s >>= 1) { if (tid < s) red[tid] += red[tid + s]; __syncthreads(); }
  float inv = 1.f / red[0];
#pragma unroll
  for (int t = 0; t < 4; ++t) {
    int i = tid + t * 256;
    if (i < PIX)       ymn[i] = (_Float16)(v[t] * inv);
    else if (i < KPAD) ymn[i] = (_Float16)0.f;   // zero K-padding
  }
}

// Exact align-corners 30->59 upsample (even: copy, odd: average) into the
// second channel half of the output.
__global__ void upsample_kernel(const float* __restrict__ xp, float* __restrict__ out) {
  long long idx = (long long)blockIdx.x * blockDim.x + threadIdx.x;
  const long long total = (long long)N_IMG * C_IN * HWFULL;
  if (idx >= total) return;
  int p = (int)(idx % HWFULL);
  int c = (int)((idx / HWFULL) % C_IN);
  int n = (int)(idx / ((long long)HWFULL * C_IN));
  int u = p / WFULL, vq = p % WFULL;
  const float* Xp = xp + ((long long)n * C_IN + c) * PIX;
  int iu = u >> 1, iv = vq >> 1;
  bool ou = u & 1, ov = vq & 1;
  float val;
  if (!ou && !ov) val = Xp[iu * WS + iv];
  else if (!ou)   val = 0.5f * (Xp[iu * WS + iv] + Xp[iu * WS + iv + 1]);
  else if (!ov)   val = 0.5f * (Xp[iu * WS + iv] + Xp[(iu + 1) * WS + iv]);
  else            val = 0.25f * (Xp[iu * WS + iv] + Xp[iu * WS + iv + 1] +
                                 Xp[(iu + 1) * WS + iv] + Xp[(iu + 1) * WS + iv + 1]);
  out[((long long)n * (2 * C_IN) + C_IN + c) * HWFULL + p] = val;
}

// ---------------- launch ----------------------------------------------------
extern "C" void kernel_launch(void* const* d_in, const int* in_sizes, int n_in,
                              void* d_out, int out_size, void* d_ws, size_t ws_size,
                              hipStream_t stream) {
  (void)in_sizes; (void)n_in; (void)out_size; (void)ws_size;
  const float* x    = (const float*)d_in[0];
  const float* rw   = (const float*)d_in[1];
  const float* rbn  = (const float*)d_in[2];
  const float* a1w  = (const float*)d_in[3];
  const float* abn  = (const float*)d_in[4];
  const float* a2w  = (const float*)d_in[5];
  const float* rpw  = (const float*)d_in[6];
  const float* rpbn = (const float*)d_in[7];
  const float* ap1w = (const float*)d_in[8];
  const float* apbn = (const float*)d_in[9];
  const float* ap2w = (const float*)d_in[10];
  const float* pjw  = (const float*)d_in[11];
  const float* pjbn = (const float*)d_in[12];
  float* out = (float*)d_out;

  char* ws = (char*)d_ws;
  size_t off = 0;
  auto alloc = [&](size_t bytes) { void* p = ws + off; off = (off + bytes + 255) & ~(size_t)255; return p; };

  _Float16* wRW   = (_Float16*)alloc((size_t)MIDC * C_IN * 2);
  _Float16* wRPW  = (_Float16*)alloc((size_t)MIDC * C_IN * 2);
  _Float16* wA1   = (_Float16*)alloc((size_t)MIDC * MIDC * 2);
  _Float16* wAP1  = (_Float16*)alloc((size_t)MIDC * MIDC * 2);
  _Float16* wA2   = (_Float16*)alloc((size_t)MMASK * MIDC * 2);
  _Float16* wAP2  = (_Float16*)alloc((size_t)MMASK * MIDC * 2);
  _Float16* wPJ   = (_Float16*)alloc((size_t)C_IN * (2 * MIDC) * 2);
  _Float16* XS    = (_Float16*)alloc((size_t)N_IMG * PIX * C_IN * 2);
  _Float16* XCcol = (_Float16*)alloc((size_t)N_IMG * PIX * MIDC * 2);
  _Float16* XCrow = (_Float16*)alloc((size_t)N_IMG * MIDC * PIX * 2 + 64); // +slack for K-pad over-read
  _Float16* Zcol  = (_Float16*)alloc((size_t)N_IMG * PIX * MIDC * 2);
  float*    Ybuf  = (float*)   alloc((size_t)N_IMG * MMASK * PIX * 4);
  _Float16* YM    = (_Float16*)alloc((size_t)N_IMG * PIX * KPAD * 2);
  _Float16* XCAT  = (_Float16*)alloc((size_t)N_IMG * PIX * (2 * MIDC) * 2);
  float*    XP    = (float*)   alloc((size_t)N_IMG * C_IN * PIX * 4);

  // 1) weights -> fp16 (row-major (out,in) already matches A layout)
  struct { const float* s; _Float16* d; long long n; } cv[7] = {
    {rw,  wRW,  (long long)MIDC * C_IN}, {rpw, wRPW, (long long)MIDC * C_IN},
    {a1w, wA1,  (long long)MIDC * MIDC}, {ap1w, wAP1, (long long)MIDC * MIDC},
    {a2w, wA2,  (long long)MMASK * MIDC}, {ap2w, wAP2, (long long)MMASK * MIDC},
    {pjw, wPJ,  (long long)C_IN * 2 * MIDC}};
  for (int i = 0; i < 7; ++i)
    f32_to_f16_kernel<<<dim3((unsigned)((cv[i].n + 255) / 256)), 256, 0, stream>>>(cv[i].s, cv[i].d, cv[i].n);

  // 2) subsampled, fp16, col-major activations
  {
    long long tot = (long long)N_IMG * PIX * C_IN;
    pack_x_kernel<<<dim3((unsigned)((tot + 255) / 256)), 256, 0, stream>>>(x, XS);
  }

  const int nTile = (PIX + 15) / 16;                       // 57
  const unsigned gy = (nTile + 8 * NTPW - 1) / (8 * NTPW); // 2 (block = 8 waves x 4 tiles)
  for (int br = 0; br < 2; ++br) {
    const _Float16* w1 = br ? wRPW : wRW;
    const _Float16* w2 = br ? wAP1 : wA1;
    const _Float16* w3 = br ? wAP2 : wA2;
    const float* bn1 = br ? rpbn : rbn;
    const float* bn2 = br ? apbn : abn;

    // GEMM1: (512x2048)*(2048x900) + BN + ReLU -> col fp16 (for GEMM2) and row fp16 (for agg)
    gemm_wmma_kernel<<<dim3(MIDC / 16, gy, N_IMG), 256, 0, stream>>>(
        w1, XS, nullptr, XCcol, XCrow, bn1, 1,
        MIDC, PIX, C_IN, C_IN, C_IN / 32,
        0LL, (long long)PIX * C_IN,
        0, 0LL, MIDC, (long long)PIX * MIDC, PIX, (long long)MIDC * PIX);

    // GEMM2: (512x512)*(512x900) + BN + ReLU
    gemm_wmma_kernel<<<dim3(MIDC / 16, gy, N_IMG), 256, 0, stream>>>(
        w2, XCcol, nullptr, Zcol, nullptr, bn2, 1,
        MIDC, PIX, MIDC, MIDC, MIDC / 32,
        0LL, (long long)PIX * MIDC,
        0, 0LL, MIDC, (long long)PIX * MIDC, 0, 0LL);

    // GEMM3: (3481x512)*(512x900) -> fp32 mask logits
    gemm_wmma_kernel<<<dim3((MMASK + 15) / 16, gy, N_IMG), 256, 0, stream>>>(
        w3, Zcol, Ybuf, nullptr, nullptr, nullptr, 0,
        MMASK, PIX, MIDC, MIDC, MIDC / 32,
        0LL, (long long)PIX * MIDC,
        PIX, (long long)MMASK * PIX, 0, 0LL, 0, 0LL);

    // PSA gather + softmax -> ym fp16 col-major, K zero-padded to 928
    if (br == 0) psa_softmax_kernel<1><<<dim3(PIX, N_IMG), 256, 0, stream>>>(Ybuf, YM);
    else         psa_softmax_kernel<0><<<dim3(PIX, N_IMG), 256, 0, stream>>>(Ybuf, YM);

    // agg: (512x900)*(900x900) -> rows [br*512, br*512+512) of XCAT (K padded to 928)
    gemm_wmma_kernel<<<dim3(MIDC / 16, gy, N_IMG), 256, 0, stream>>>(
        XCrow, YM, nullptr, XCAT + br * MIDC, nullptr, nullptr, 0,
        MIDC, PIX, PIX, KPAD, KPAD / 32,
        (long long)MIDC * PIX, (long long)PIX * KPAD,
        0, 0LL, 2 * MIDC, (long long)PIX * (2 * MIDC), 0, 0LL);
  }

  // GEMM4: (2048x1024)*(1024x900) + BN + ReLU -> fp32
  gemm_wmma_kernel<<<dim3(C_IN / 16, gy, N_IMG), 256, 0, stream>>>(
      wPJ, XCAT, XP, nullptr, nullptr, pjbn, 1,
      C_IN, PIX, 2 * MIDC, 2 * MIDC, (2 * MIDC) / 32,
      0LL, (long long)PIX * 2 * MIDC,
      PIX, (long long)C_IN * PIX, 0, 0LL, 0, 0LL);

  // concat: first 2048 channels per image are a straight copy of x
  for (int n = 0; n < N_IMG; ++n)
    hipMemcpyAsync(out + (long long)n * (2 * C_IN) * HWFULL,
                   x + (long long)n * C_IN * HWFULL,
                   (size_t)C_IN * HWFULL * sizeof(float), hipMemcpyDeviceToDevice, stream);

  // exact 30->59 upsample into second channel half
  {
    long long tot = (long long)N_IMG * C_IN * HWFULL;
    upsample_kernel<<<dim3((unsigned)((tot + 255) / 256)), 256, 0, stream>>>(XP, out);
  }
}
